// GeometryAwareTensorAttention_20907900797506
// MI455X (gfx1250) — matrix operand order repaired and checked by reading, not compiled, not run on hardware
//
#include <hip/hip_runtime.h>

typedef unsigned short u16;
typedef __attribute__((ext_vector_type(16))) __bf16 v16bf;
typedef __attribute__((ext_vector_type(8)))  float  v8f;

struct alignas(16) U128 { unsigned int w[4]; };
union FragBF { U128 u[2]; v16bf v; };

__device__ __forceinline__ u16 f2bf(float f) {
  union { float f; unsigned int u; } x; x.f = f;
  unsigned int r = x.u + 0x7FFFu + ((x.u >> 16) & 1u);
  return (u16)(r >> 16);
}

__device__ __forceinline__ v8f wmma_bf16(const FragBF& a, const FragBF& b, v8f c) {
  return __builtin_amdgcn_wmma_f32_16x16x32_bf16(false, a.v, false, b.v, (short)0, c,
                                                 false, false);
}

// ---------------------------------------------------------------- LayerNorm
__global__ void k_ln(const float* __restrict__ h, const float* __restrict__ g_hi,
                     const float* __restrict__ g_hj, u16* __restrict__ hi_bf,
                     u16* __restrict__ hj_bf) {
  __shared__ float red[64];
  int i = blockIdx.x, c = threadIdx.x;
  float x = h[i * 64 + c];
  red[c] = x; __syncthreads();
  for (int o = 32; o > 0; o >>= 1) { if (c < o) red[c] += red[c + o]; __syncthreads(); }
  float mean = red[0] * (1.f / 64.f);
  __syncthreads();
  float xm = x - mean;
  red[c] = xm * xm; __syncthreads();
  for (int o = 32; o > 0; o >>= 1) { if (c < o) red[c] += red[c + o]; __syncthreads(); }
  float inv = rsqrtf(red[0] * (1.f / 64.f) + 1e-5f);
  float yn = xm * inv;
  hi_bf[i * 64 + c] = f2bf(yn * g_hi[c]);
  hj_bf[i * 64 + c] = f2bf(yn * g_hj[c]);
}

// ------------------------------------------------ f32 -> bf16 (same layout)
__global__ void k_conv(const float* __restrict__ in, u16* __restrict__ out, int n) {
  for (int idx = blockIdx.x * blockDim.x + threadIdx.x; idx < n;
       idx += gridDim.x * blockDim.x)
    out[idx] = f2bf(in[idx]);
}

// ------------------------------------- f32 (K x N) -> bf16 transposed (N x K)
__global__ void k_convT(const float* __restrict__ in, u16* __restrict__ out,
                        int K, int N) {
  int total = K * N;
  for (int idx = blockIdx.x * blockDim.x + threadIdx.x; idx < total;
       idx += gridDim.x * blockDim.x) {
    int k = idx / N, n = idx - k * N;
    out[n * K + k] = f2bf(in[idx]);
  }
}

// ------------------------------------------------------------- bf16 WMMA GEMM
// C[MxN] = act(A[MxK] @ B + bias), B given transposed (BT rows = N, cols = K).
// One wave computes one 16x16 tile; 8 waves per block.
__global__ void k_gemm(const u16* __restrict__ A, int lda,
                       const u16* __restrict__ BT, int ldb,
                       const float* __restrict__ bias,
                       float* __restrict__ C, int ldc,
                       u16* __restrict__ Cbf, int ldcbf,
                       int M, int N, int K, int act) {
  int wave = threadIdx.x >> 5, lane = threadIdx.x & 31;
  int hi = lane >> 4, lo = lane & 15;
  int ntN = N >> 4;
  int tiles = (M >> 4) * ntN;
  int tile = blockIdx.x * (blockDim.x >> 5) + wave;
  if (tile >= tiles) return;
  int m0 = (tile / ntN) << 4, n0 = (tile % ntN) << 4;

  const u16* Arow = A + (size_t)(m0 + lo) * lda;
  const u16* Brow = BT + (size_t)(n0 + lo) * ldb;
  v8f acc = {};
  for (int kb = 0; kb < K; kb += 32) {
    __builtin_prefetch(Brow + kb + 32, 0, 0);     // global_prefetch_b8
    FragBF a, b;
    int ab = kb + hi * 8;
    a.u[0] = *(const U128*)(Arow + ab);
    a.u[1] = *(const U128*)(Arow + ab + 16);
    int bb = kb + hi * 8;
    b.u[0] = *(const U128*)(Brow + bb);
    b.u[1] = *(const U128*)(Brow + bb + 16);
    acc = wmma_bf16(a, b, acc);
  }
  int n = n0 + lo;
  float bv = bias ? bias[n] : 0.f;
#pragma unroll
  for (int r = 0; r < 8; ++r) {
    int m = m0 + r + (hi << 3);
    float v = acc[r] + bv;
    if (act) v = v / (1.f + expf(-v));           // silu
    C[(size_t)m * ldc + n] = v;
    if (Cbf) Cbf[(size_t)m * ldcbf + n] = f2bf(v);
  }
}

// -------------------------------------------------- Tsum[i,c] = sum_j t[i,j,c]
__global__ void k_tsum(const float* __restrict__ t, float* __restrict__ tsum) {
  int i = blockIdx.x, c = threadIdx.x;
  float acc = 0.f;
  const float* p = t + (size_t)i * 128 * 64 + c;
  for (int j = 0; j < 128; ++j) acc += p[j * 64];
  tsum[i * 64 + c] = acc;
}

// sim_den[h,i,s] = sum_c Tsum[i,c] * sum_d q[i,h*64+d] * Wev[c, s*512+h*64+d]
__global__ void k_simden(const float* __restrict__ qf, const float* __restrict__ Wev,
                         const float* __restrict__ tsum, float* __restrict__ simden) {
  __shared__ float red[64];
  int blk = blockIdx.x;            // h*128 + i
  int h = blk >> 7, i = blk & 127;
  int c = threadIdx.x;
  const float* qrow = qf + (size_t)i * 512 + h * 64;
  float tv = tsum[i * 64 + c];
  for (int s = 0; s < 5; ++s) {
    const float* wrow = Wev + (size_t)c * 2560 + s * 512 + h * 64;
    float acc = 0.f;
#pragma unroll
    for (int d = 0; d < 64; ++d) acc += qrow[d] * wrow[d];
    red[c] = tv * acc; __syncthreads();
    for (int o = 32; o > 0; o >>= 1) { if (c < o) red[c] += red[c + o]; __syncthreads(); }
    if (c == 0) simden[(size_t)blk * 5 + s] = red[0];
    __syncthreads();
  }
}

// ---------------------------------------------------------- fused ev/sea/out
// Block = (j-tile of 16, i, s), 128 threads (4 waves).
// Phase 1: ev tile (16 x 512) via WMMA (t rows x WevT), fused attn*vals+ev*pav,
//          bf16 result staged to LDS.
// Phase 2: out tile (16 x 64) = Asea(16x512) @ Wc, A read from LDS.
__global__ void k_out(const u16* __restrict__ t_bf, const u16* __restrict__ WevT,
                      const u16* __restrict__ WcT,
                      const float* __restrict__ simnum,
                      const float* __restrict__ simden,
                      const float* __restrict__ vals, const float* __restrict__ pav,
                      float* __restrict__ outb) {
  __shared__ u16 asea[16 * 512];
  int j0 = blockIdx.x << 4, i = blockIdx.y, s = blockIdx.z;
  int wave = threadIdx.x >> 5, lane = threadIdx.x & 31;
  int hi = lane >> 4, lo = lane & 15;

  // A fragments: t rows (K = 64 -> two k-steps), shared by all 8 n-tiles.
  const u16* Arow = t_bf + ((size_t)(i * 128 + j0 + lo)) * 64;
  FragBF a0, a1;
  a0.u[0] = *(const U128*)(Arow + hi * 8);
  a0.u[1] = *(const U128*)(Arow + hi * 8 + 16);
  a1.u[0] = *(const U128*)(Arow + 32 + hi * 8);
  a1.u[1] = *(const U128*)(Arow + 32 + hi * 8 + 16);

  for (int nt = 0; nt < 8; ++nt) {
    int f0 = wave * 128 + nt * 16;                 // this wave's f-chunk
    const u16* Brow = WevT + ((size_t)(s * 512 + f0 + lo)) * 64;
    FragBF b0, b1;
    b0.u[0] = *(const U128*)(Brow + hi * 8);
    b0.u[1] = *(const U128*)(Brow + hi * 8 + 16);
    b1.u[0] = *(const U128*)(Brow + 32 + hi * 8);
    b1.u[1] = *(const U128*)(Brow + 32 + hi * 8 + 16);
    v8f ev = {};
    ev = wmma_bf16(a0, b0, ev);
    ev = wmma_bf16(a1, b1, ev);

    int h = f0 >> 6;                               // head for this 16-col tile
    float rden = 1.f / simden[(size_t)(h * 128 + i) * 5 + s];
    int fl = f0 + lo;
    const float* vcol = vals + s * 512 + fl;
    const float* pcol = pav  + s * 512 + fl;
    const float* snrow = simnum + (size_t)(h * 128 + i) * 128 + j0;
#pragma unroll
    for (int r = 0; r < 8; ++r) {
      int m = r + (hi << 3);
      int j = j0 + m;
      float attn = snrow[m] * rden;
      float as = attn * vcol[(size_t)j * 2560] + ev[r] * pcol[(size_t)j * 2560];
      asea[m * 512 + fl] = f2bf(as);
    }
  }
  __syncthreads();

  // Phase 2: each wave -> 16 output columns e in [wave*16, wave*16+16)
  v8f acc = {};
  const u16* Brow2 = WcT + (size_t)(wave * 16 + lo) * 512;
  for (int kb = 0; kb < 512; kb += 32) {
    FragBF a, b;
    const u16* Al = &asea[lo * 512 + kb + hi * 8];
    a.u[0] = *(const U128*)(Al);
    a.u[1] = *(const U128*)(Al + 16);
    b.u[0] = *(const U128*)(Brow2 + kb + hi * 8);
    b.u[1] = *(const U128*)(Brow2 + kb + hi * 8 + 16);
    acc = wmma_bf16(a, b, acc);
  }
  int e = wave * 16 + lo;
#pragma unroll
  for (int r = 0; r < 8; ++r) {
    int m = r + (hi << 3);
    int j = j0 + m;
    outb[(((size_t)i * 128 + j) * 5 + s) * 64 + e] = acc[r];
  }
}

// ------------------------------------------- h_new[i,e] = h + sum_j out[i,j,0,e]
__global__ void k_hnew(const float* __restrict__ h, const float* __restrict__ outb,
                       float* __restrict__ d_out) {
  int i = blockIdx.x, e = threadIdx.x;
  float acc = h[i * 64 + e];
  const float* p = outb + ((size_t)i * 128) * 5 * 64 + e;   // s=0 slice
  for (int j = 0; j < 128; ++j) acc += p[(size_t)j * 5 * 64];
  d_out[i * 64 + e] = acc;
}

// x_new0[i,d,m] = sum_j x1[j,d,m]*out[i,j,3,d] + r1[i,j,m]*out[i,j,1,d]  (m<3)
// x_new1[i,d,m] = sum_j x2[j,d,m]*out[i,j,4,d] + r2[i,j,m]*out[i,j,2,d]  (m<5)
__global__ void k_xnew(const float* __restrict__ outb,
                       const float* __restrict__ r1, const float* __restrict__ r2,
                       const float* __restrict__ x1, const float* __restrict__ x2,
                       float* __restrict__ d_out) {
  int i = blockIdx.x, d = threadIdx.x;
  float a0[3] = {0.f, 0.f, 0.f};
  float a1[5] = {0.f, 0.f, 0.f, 0.f, 0.f};
  for (int j = 0; j < 128; ++j) {
    const float* ob = outb + (((size_t)i * 128 + j) * 5) * 64 + d;
    float o1 = ob[1 * 64], o2 = ob[2 * 64], o3 = ob[3 * 64], o4 = ob[4 * 64];
    const float* r1p = r1 + (size_t)(i * 128 + j) * 3;
    const float* r2p = r2 + (size_t)(i * 128 + j) * 5;
    const float* x1p = x1 + (size_t)(j * 64 + d) * 3;
    const float* x2p = x2 + (size_t)(j * 64 + d) * 5;
#pragma unroll
    for (int m = 0; m < 3; ++m) a0[m] += x1p[m] * o3 + r1p[m] * o1;
#pragma unroll
    for (int m = 0; m < 5; ++m) a1[m] += x2p[m] * o4 + r2p[m] * o2;
  }
  float* o0 = d_out + 8192 + (size_t)(i * 64 + d) * 3;
  float* o1 = d_out + 8192 + 24576 + (size_t)(i * 64 + d) * 5;
#pragma unroll
  for (int m = 0; m < 3; ++m) o0[m] = a0[m];
#pragma unroll
  for (int m = 0; m < 5; ++m) o1[m] = a1[m];
}

// ===========================================================================
extern "C" void kernel_launch(void* const* d_in, const int* in_sizes, int n_in,
                              void* d_out, int out_size, void* d_ws, size_t ws_size,
                              hipStream_t stream) {
  const float* h    = (const float*)d_in[0];
  const float* tij  = (const float*)d_in[1];
  const float* r1   = (const float*)d_in[2];
  const float* r2   = (const float*)d_in[3];
  const float* x1   = (const float*)d_in[4];
  const float* x2   = (const float*)d_in[5];
  const float* g_hi = (const float*)d_in[6];
  const float* g_hj = (const float*)d_in[7];
  const float* Wq   = (const float*)d_in[8];
  const float* Wk   = (const float*)d_in[9];
  const float* Wv1  = (const float*)d_in[10];
  const float* bv1  = (const float*)d_in[11];
  const float* Wv2  = (const float*)d_in[12];
  const float* bv2  = (const float*)d_in[13];
  const float* Wp1  = (const float*)d_in[14];
  const float* bp1  = (const float*)d_in[15];
  const float* Wp2  = (const float*)d_in[16];
  const float* bp2  = (const float*)d_in[17];
  const float* Wev  = (const float*)d_in[18];
  const float* Wc   = (const float*)d_in[19];
  float* out = (float*)d_out;

  char* ws = (char*)d_ws;
  size_t off = 0;
  auto take = [&](size_t bytes) -> void* {
    void* p = ws + off;
    off = (off + bytes + 255) & ~(size_t)255;
    return p;
  };
  u16* hi_bf = (u16*)take(128 * 64 * 2);
  u16* hj_bf = (u16*)take(128 * 64 * 2);
  u16* WqT   = (u16*)take(512 * 64 * 2);
  u16* WkT   = (u16*)take(512 * 64 * 2);
  u16* Wv1T  = (u16*)take(1024 * 64 * 2);
  u16* Wp1T  = (u16*)take(1024 * 64 * 2);
  u16* Wv2T  = (u16*)take((size_t)2560 * 1024 * 2);
  u16* Wp2T  = (u16*)take((size_t)2560 * 1024 * 2);
  u16* WevT  = (u16*)take(2560 * 64 * 2);
  u16* WcT   = (u16*)take(64 * 512 * 2);
  u16* t_bf  = (u16*)take((size_t)128 * 128 * 64 * 2);
  float* qf  = (float*)take(128 * 512 * 4);
  u16*   qbf = (u16*)take(128 * 512 * 2);
  float* kf  = (float*)take(128 * 512 * 4);
  u16*   kbf = (u16*)take(128 * 512 * 2);
  float* t1f = (float*)take(128 * 1024 * 4);
  u16*   t1bf= (u16*)take(128 * 1024 * 2);
  float* p1f = (float*)take(128 * 1024 * 4);
  u16*   p1bf= (u16*)take(128 * 1024 * 2);
  float* vals   = (float*)take(128 * 2560 * 4);
  float* pavb   = (float*)take(128 * 2560 * 4);
  float* simnum = (float*)take(8 * 128 * 128 * 4);
  float* tsum   = (float*)take(128 * 64 * 4);
  float* simden = (float*)take(8 * 128 * 5 * 4);
  float* outb   = (float*)take((size_t)128 * 128 * 5 * 64 * 4);
  (void)ws_size; (void)n_in; (void)in_sizes; (void)out_size;

  // 1. LayerNorms -> bf16
  k_ln<<<128, 64, 0, stream>>>(h, g_hi, g_hj, hi_bf, hj_bf);

  // 2. Convert / transpose weights & t_ij to bf16
  k_conv<<<1024, 256, 0, stream>>>(tij, t_bf, 128 * 128 * 64);
  k_convT<<<128, 256, 0, stream>>>(Wq, WqT, 64, 512);
  k_convT<<<128, 256, 0, stream>>>(Wk, WkT, 64, 512);
  k_convT<<<256, 256, 0, stream>>>(Wv1, Wv1T, 64, 1024);
  k_convT<<<256, 256, 0, stream>>>(Wp1, Wp1T, 64, 1024);
  k_convT<<<4096, 256, 0, stream>>>(Wv2, Wv2T, 1024, 2560);
  k_convT<<<4096, 256, 0, stream>>>(Wp2, Wp2T, 1024, 2560);
  k_convT<<<640, 256, 0, stream>>>(Wev, WevT, 64, 2560);
  k_convT<<<128, 256, 0, stream>>>(Wc, WcT, 512, 64);

  // 3. Projections (WMMA GEMMs)
  k_gemm<<<32, 256, 0, stream>>>(hi_bf, 64, WqT, 64, nullptr, qf, 512, qbf, 512,
                                 128, 512, 64, 0);
  k_gemm<<<32, 256, 0, stream>>>(hj_bf, 64, WkT, 64, nullptr, kf, 512, kbf, 512,
                                 128, 512, 64, 0);
  k_gemm<<<64, 256, 0, stream>>>(hj_bf, 64, Wv1T, 64, bv1, t1f, 1024, t1bf, 1024,
                                 128, 1024, 64, 1);
  k_gemm<<<64, 256, 0, stream>>>(hj_bf, 64, Wp1T, 64, bp1, p1f, 1024, p1bf, 1024,
                                 128, 1024, 64, 1);
  k_gemm<<<160, 256, 0, stream>>>(t1bf, 1024, Wv2T, 1024, bv2, vals, 2560,
                                  (u16*)nullptr, 0, 128, 2560, 1024, 0);
  k_gemm<<<160, 256, 0, stream>>>(p1bf, 1024, Wp2T, 1024, bp2, pavb, 2560,
                                  (u16*)nullptr, 0, 128, 2560, 1024, 0);

  // 4. sim_num per head: q_h (128x64) @ k_h^T
  for (int hh = 0; hh < 8; ++hh) {
    k_gemm<<<8, 256, 0, stream>>>(qbf + hh * 64, 512, kbf + hh * 64, 512, nullptr,
                                  simnum + (size_t)hh * 128 * 128, 128,
                                  (u16*)nullptr, 0, 128, 128, 64, 0);
  }

  // 5. sim_den via separable contraction (no ev materialization)
  k_tsum<<<128, 64, 0, stream>>>(tij, tsum);
  k_simden<<<1024, 64, 0, stream>>>(qf, Wev, tsum, simden);

  // 6. Fused ev -> sea -> out (two chained WMMA GEMMs through LDS)
  k_out<<<dim3(8, 128, 5), 128, 0, stream>>>(t_bf, WevT, WcT, simnum, simden,
                                             vals, pavb, outb);

  // 7. Final reductions over j
  k_hnew<<<128, 64, 0, stream>>>(h, outb, out);
  k_xnew<<<128, 64, 0, stream>>>(outb, r1, r2, x1, x2, out);
}
